// WordnnEmbedding_73461120631068
// MI455X (gfx1250) — matrix operand order repaired and checked by reading, not compiled, not run on hardware
//
#include <hip/hip_runtime.h>
#include <hip/hip_bf16.h>

typedef __attribute__((ext_vector_type(16))) _Float16 v16h;
typedef __attribute__((ext_vector_type(8)))  float    v8f;
typedef __attribute__((ext_vector_type(4)))  float    f4;
typedef __attribute__((ext_vector_type(4)))  int      i4;

// Problem constants (from reference)
constexpr int Bn    = 2;
constexpr int Hh    = 512;
constexpr int Wd    = 512;
constexpr int Nb    = 200;
constexpr int VOCAB = 30552;
constexpr int HID   = 768;
constexpr int EDIM  = 64;
constexpr int NTOK  = Bn * Nb + 1;          // 401 (incl. background token id 0)
constexpr int MT    = (NTOK + 15) & ~15;    // 416 rows (26 M-tiles of 16)

// ---------------------------------------------------------------------------
// Phase 1: P[t, d] = sum_c emb_table[id_t, c] * proj_w[d, c]
// One wave per 16x16 output tile; wave w in block handles N-tile w (N=64 -> 4 waves).
// K = 768 swept in 24 steps of 32 via v_wmma_f32_16x16x32_f16.
// ---------------------------------------------------------------------------
__global__ __launch_bounds__(128)
void tok_proj_wmma(const long long* __restrict__ ids,   // [Bn*Nb] int64 token ids
                   const float*     __restrict__ emb,   // [VOCAB, HID]
                   const float*     __restrict__ pw,    // [EDIM, HID]
                   float*           __restrict__ P)     // [MT, EDIM]
{
    const int lane = threadIdx.x & 31;
    const int wv   = threadIdx.x >> 5;   // N-tile index (0..3)
    const int mt   = blockIdx.x;         // M-tile index (0..25)
    const int lm   = lane & 15;
    const int g    = lane >> 4;          // lane group (0: lanes 0-15, 1: lanes 16-31)

    // A-matrix row: token id for row (mt*16 + lm); pad & background use id 0.
    const int t  = mt * 16 + lm;
    int id = 0;
    if (t < Bn * Nb) id = (int)ids[t];
    if (id < 0) id = 0;
    if (id >= VOCAB) id = VOCAB - 1;
    const float* __restrict__ arow = emb + (long)id * HID;
    // B-matrix column: proj_w row d = wv*16 + lm
    const float* __restrict__ brow = pw + (long)(wv * 16 + lm) * HID;

    v8f c = {};
    #pragma unroll 4
    for (int k0 = 0; k0 < HID; k0 += 32) {
        // --- A fragment (16x32 f16): halves 0..7  -> K = k0 + 8*g + j
        //                             halves 8..15 -> K = k0 + 16 + 8*g + (j-8)
        float af[16];
        *(f4*)(af + 0)  = *(const f4*)(arow + k0 + 8 * g);
        *(f4*)(af + 4)  = *(const f4*)(arow + k0 + 8 * g + 4);
        *(f4*)(af + 8)  = *(const f4*)(arow + k0 + 16 + 8 * g);
        *(f4*)(af + 12) = *(const f4*)(arow + k0 + 16 + 8 * g + 4);
        // --- B fragment (32x16 f16): same K mapping as A, N = lm
        float bf[16];
        *(f4*)(bf + 0)  = *(const f4*)(brow + k0 + 8 * g);
        *(f4*)(bf + 4)  = *(const f4*)(brow + k0 + 8 * g + 4);
        *(f4*)(bf + 8)  = *(const f4*)(brow + k0 + 16 + 8 * g);
        *(f4*)(bf + 12) = *(const f4*)(brow + k0 + 16 + 8 * g + 4);

        v16h a, b;
        #pragma unroll
        for (int j = 0; j < 16; ++j) { a[j] = (_Float16)af[j]; b[j] = (_Float16)bf[j]; }

        // D = A x B + C   (f32 accumulate)
        c = __builtin_amdgcn_wmma_f32_16x16x32_f16(
                /*neg_a=*/false, a, /*neg_b=*/false, b,
                /*c_mod=*/(short)0, c, /*reuse_a=*/false, /*reuse_b=*/false);
    }

    // D layout: lane group g, VGPR r -> (M = r + 8*g, N = lm)
    float* __restrict__ outp = P + (mt * 16 + 8 * g) * EDIM + wv * 16 + lm;
    #pragma unroll
    for (int r = 0; r < 8; ++r) outp[r * EDIM] = c[r];
}

// ---------------------------------------------------------------------------
// Phase 2: rasterize — per pixel, last (max-index) covering box -> token index.
// Block = 256 consecutive pixels (same b since H*W % 256 == 0); bboxes in LDS.
// ---------------------------------------------------------------------------
__global__ __launch_bounds__(256)
void rasterize_kernel(const int* __restrict__ bbox,     // [Bn, Nb, 4] (ws,hs,we,he)
                      const int* __restrict__ stridep,  // scalar
                      int*       __restrict__ gidx)     // [Bn*Hh*Wd] token index
{
    __shared__ int sb[Nb * 4];
    const int pix = blockIdx.x * 256 + threadIdx.x;
    const int b   = pix / (Hh * Wd);

    const float inv = 1.0f / (float)stridep[0];
    const int* __restrict__ src = bbox + b * Nb * 4;
    for (int i = threadIdx.x; i < Nb * 4; i += 256)
        sb[i] = (int)rintf((float)src[i] * inv);
    __syncthreads();

    const int hw = pix % (Hh * Wd);
    const int h  = hw / Wd;
    const int w  = hw % Wd;

    int best = -1;
    #pragma unroll 4
    for (int n = 0; n < Nb; ++n) {
        const int ws_ = sb[n * 4 + 0];
        const int hs  = sb[n * 4 + 1];
        const int we  = sb[n * 4 + 2];
        const int he  = sb[n * 4 + 3];
        const bool in = (h >= hs) & (h < he) & (w >= ws_) & (w < we);
        if (in) best = n;
    }
    gidx[pix] = (best >= 0) ? (b * Nb + best) : (Bn * Nb);  // 400 = background
}

// ---------------------------------------------------------------------------
// Phase 3: scatter — out[b, d, h, w] = P[gidx[b,h,w], d].
// 4-wide along w: int4 gidx load, 4 cached P reads (106 KB, L2-resident),
// non-temporal b128 output store (write-once stream, keep it out of L2).
// ---------------------------------------------------------------------------
__global__ __launch_bounds__(256)
void scatter_kernel(const int*   __restrict__ gidx,
                    const float* __restrict__ P,
                    float*       __restrict__ out)
{
    const int q   = blockIdx.x * 256 + threadIdx.x;     // vec4 index, < 2^23
    const int o   = q * 4;                              // element index
    const int HW  = Hh * Wd;
    const int b   = o / (EDIM * HW);
    const int rem = o % (EDIM * HW);
    const int d   = rem / HW;
    const int hw  = rem % HW;                           // multiple of 4

    const i4 t = *(const i4*)(gidx + b * HW + hw);
    f4 v;
    v.x = P[t.x * EDIM + d];
    v.y = P[t.y * EDIM + d];
    v.z = P[t.z * EDIM + d];
    v.w = P[t.w * EDIM + d];
    __builtin_nontemporal_store(v, (f4*)(out + o));
}

// ---------------------------------------------------------------------------
// Inputs (setup_inputs order): img f32 (unused), input_ids i64, bbox i32,
// stride scalar, emb_table f32, proj_w f32. Output: f32 [B, 64, H, W].
// ---------------------------------------------------------------------------
extern "C" void kernel_launch(void* const* d_in, const int* in_sizes, int n_in,
                              void* d_out, int out_size, void* d_ws, size_t ws_size,
                              hipStream_t stream) {
    const long long* ids     = (const long long*)d_in[1];
    const int*       bbox    = (const int*)d_in[2];
    const int*       stridep = (const int*)d_in[3];
    const float*     emb     = (const float*)d_in[4];
    const float*     pw      = (const float*)d_in[5];
    float*           out     = (float*)d_out;

    // Workspace: P [416*64 f32] then gidx [B*H*W int] (256B aligned) ≈ 2.2 MB
    float* P    = (float*)d_ws;
    size_t poff = ((size_t)MT * EDIM * sizeof(float) + 255) & ~(size_t)255;
    int*   gidx = (int*)((char*)d_ws + poff);

    tok_proj_wmma<<<MT / 16, 128, 0, stream>>>(ids, emb, pw, P);
    rasterize_kernel<<<(Bn * Hh * Wd) / 256, 256, 0, stream>>>(bbox, stridep, gidx);
    scatter_kernel<<<(Bn * EDIM * Hh * Wd) / (256 * 4), 256, 0, stream>>>(gidx, P, out);
}
